// Transformer_26448408608861
// MI455X (gfx1250) — compile-verified, matrix operand order checked by reading, not run on hardware
//
#include <hip/hip_runtime.h>
#include <hip/hip_bf16.h>
#include <math.h>

// ---------------------------------------------------------------------------
// Model constants (match reference)
// ---------------------------------------------------------------------------
constexpr int BB   = 4;
constexpr int TT   = 1024;
constexpr int DD   = 1024;
constexpr int HH   = 16;
constexpr int HD   = DD / HH;     // 64
constexpr int VV   = 32000;
constexpr int NBLK = 2;
constexpr int DFF  = 4 * DD;      // 4096
constexpr int MM   = BB * TT;     // 4096 rows
constexpr float EPSF = 1e-5f;

// GEMM tiling
constexpr int BM = 128, BN = 128, BK = 32;
constexpr int LDSS = 40;          // BK + 8 pad elems (80B rows, 16B aligned)

// ---------------------------------------------------------------------------
// WMMA vector types (gfx1250, wave32)
// ---------------------------------------------------------------------------
typedef __attribute__((ext_vector_type(16))) __bf16 v16bf;
typedef __attribute__((ext_vector_type(8)))  __bf16 v8bf;
typedef __attribute__((ext_vector_type(8)))  float  v8f;

// ---------------------------------------------------------------------------
// CDNA5 async global->LDS copy (ASYNCcnt path)
// ---------------------------------------------------------------------------
__device__ __forceinline__ uint32_t lds_off32(const void* p) {
    // generic pointer to LDS = {shared aperture, offset32}; low 32 bits = LDS byte offset
    return (uint32_t)(uintptr_t)p;
}
__device__ __forceinline__ void async_copy_b128(uint32_t lds, const void* g) {
    unsigned long long ga = (unsigned long long)g;
    asm volatile("global_load_async_to_lds_b128 %0, %1, off"
                 :: "v"(lds), "v"(ga) : "memory");
}
__device__ __forceinline__ void wait_async0() {
    asm volatile("s_wait_asynccnt 0x0" ::: "memory");
}

// ---------------------------------------------------------------------------
// Embedding: h[b,t,:] = tok_emb[x[b,t],:] + pos_emb[t,:]   (fp32 residual stream)
// ---------------------------------------------------------------------------
__global__ __launch_bounds__(256)
void embed_kernel(const int* __restrict__ x, const float* __restrict__ tok,
                  const float* __restrict__ pos, float* __restrict__ h)
{
    size_t i = (size_t)blockIdx.x * 256 + threadIdx.x;
    if (i >= (size_t)MM * DD) return;
    int d   = (int)(i % DD);
    int row = (int)(i / DD);
    int t   = row % TT;
    h[i] = tok[(size_t)x[row] * DD + d] + pos[(size_t)t * DD + d];
}

// ---------------------------------------------------------------------------
// LayerNorm: one block per row; emits bf16 (feeds GEMM A-operand only)
// ---------------------------------------------------------------------------
__global__ __launch_bounds__(256)
void layernorm_kernel(const float* __restrict__ xin, const float* __restrict__ g,
                      const float* __restrict__ bta, __bf16* __restrict__ y)
{
    __shared__ float red[256];
    const int row = blockIdx.x;
    const int tid = threadIdx.x;
    const float* xr = xin + (size_t)row * DD;

    float s = 0.f;
    for (int i = tid; i < DD; i += 256) s += xr[i];
    red[tid] = s; __syncthreads();
    for (int off = 128; off > 0; off >>= 1) {
        if (tid < off) red[tid] += red[tid + off];
        __syncthreads();
    }
    const float mean = red[0] * (1.0f / DD);
    __syncthreads();

    float v = 0.f;
    for (int i = tid; i < DD; i += 256) { float d = xr[i] - mean; v += d * d; }
    red[tid] = v; __syncthreads();
    for (int off = 128; off > 0; off >>= 1) {
        if (tid < off) red[tid] += red[tid + off];
        __syncthreads();
    }
    const float inv = rsqrtf(red[0] * (1.0f / DD) + EPSF);

    __bf16* yr = y + (size_t)row * DD;
    for (int i = tid; i < DD; i += 256)
        yr[i] = (__bf16)((xr[i] - mean) * inv * g[i] + bta[i]);
}

// ---------------------------------------------------------------------------
// Weight prep: fp32 [K,N] -> bf16 [N,K] (transpose + convert, once per launch)
// ---------------------------------------------------------------------------
__global__ __launch_bounds__(256)
void transpose_cvt_kernel(const float* __restrict__ W, __bf16* __restrict__ Wt,
                          int K, int N)
{
    __shared__ float tile[32][33];
    const int nb = blockIdx.x * 32, kb = blockIdx.y * 32;
    const int tx = threadIdx.x & 31, ty = threadIdx.x >> 5;   // 32 x 8
    #pragma unroll
    for (int i = ty; i < 32; i += 8)
        tile[i][tx] = W[(size_t)(kb + i) * N + nb + tx];
    __syncthreads();
    #pragma unroll
    for (int i = ty; i < 32; i += 8)
        Wt[(size_t)(nb + i) * K + kb + tx] = (__bf16)tile[tx][i];
}

// ---------------------------------------------------------------------------
// BF16 WMMA GEMM:  C[M,N] = epi(A[M,K] @ Bt[N,K]^T)
//   A, Bt: bf16 row-major, K contiguous.  Block tile 128x128, BK=32.
//   8 waves (4 row strips x 2 col strips); wave tile 32x64 = 2x4 wmma frags.
//   Double-buffered LDS; tiles staged with GLOBAL_LOAD_ASYNC_TO_LDS_B128 so
//   the async DMA of tile k+1 overlaps the 8 WMMAs of tile k; one barrier/iter.
//   Epilogue is compile-time configured (branch-free).
// ---------------------------------------------------------------------------
template<bool HB, bool HR, bool RELU, bool OB16>
__global__ __launch_bounds__(256)
void gemm_bf16_wmma(const __bf16* __restrict__ A, const __bf16* __restrict__ Bt,
                    const float* __restrict__ bias, const float* __restrict__ res,
                    void* __restrict__ Cv, int M, int N, int K)
{
    __shared__ __align__(16) __bf16 sA[2][BM * LDSS];
    __shared__ __align__(16) __bf16 sB[2][BN * LDSS];

    const int tid  = threadIdx.x;
    const int lane = tid & 31;
    const int wave = tid >> 5;
    const int wm   = wave >> 1;        // 0..3 : 32-row strip
    const int wn   = wave & 1;         // 0..1 : 64-col strip
    const int half = lane >> 4;        // K-half selector (ISA 16-bit A layout)
    const int l15  = lane & 15;

    const int rowBase = blockIdx.y * BM;
    const int colBase = blockIdx.x * BN;

    // per-thread staging coords: 2 chunks x (A,B), 16B each
    const int c0 = tid, c1 = tid + 256;

    auto stage = [&](int kt, int buf) {
        const uint32_t a0 = lds_off32(&sA[buf][0]);
        const uint32_t b0 = lds_off32(&sB[buf][0]);
        {
            const int r = c0 >> 2, part = (c0 & 3) * 8;
            async_copy_b128(a0 + (uint32_t)(r * LDSS + part) * 2,
                            A  + (size_t)(rowBase + r) * K + kt + part);
            async_copy_b128(b0 + (uint32_t)(r * LDSS + part) * 2,
                            Bt + (size_t)(colBase + r) * K + kt + part);
        }
        {
            const int r = c1 >> 2, part = (c1 & 3) * 8;
            async_copy_b128(a0 + (uint32_t)(r * LDSS + part) * 2,
                            A  + (size_t)(rowBase + r) * K + kt + part);
            async_copy_b128(b0 + (uint32_t)(r * LDSS + part) * 2,
                            Bt + (size_t)(colBase + r) * K + kt + part);
        }
    };

    v8f acc[2][4] = {};

    stage(0, 0);
    wait_async0();
    __syncthreads();

    int buf = 0;
    for (int kt = 0; kt < K; kt += BK) {
        // kick off DMA for the next tile into the other buffer
        if (kt + BK < K) stage(kt + BK, buf ^ 1);

        // ---- fragments: two 16B LDS loads each (ISA 16-bit operand layout) ----
        v16bf af[2], bfr[4];
        #pragma unroll
        for (int i = 0; i < 2; ++i) {
            const __bf16* p = &sA[buf][(wm * 32 + 16 * i + l15) * LDSS];
            v8bf lo = *(const v8bf*)(p + 8 * half);
            v8bf hi = *(const v8bf*)(p + 16 + 8 * half);
            af[i] = __builtin_shufflevector(lo, hi, 0,1,2,3,4,5,6,7,8,9,10,11,12,13,14,15);
        }
        #pragma unroll
        for (int j = 0; j < 4; ++j) {
            const __bf16* p = &sB[buf][(wn * 64 + 16 * j + l15) * LDSS];
            v8bf lo = *(const v8bf*)(p + 8 * half);
            v8bf hi = *(const v8bf*)(p + 16 + 8 * half);
            bfr[j] = __builtin_shufflevector(lo, hi, 0,1,2,3,4,5,6,7,8,9,10,11,12,13,14,15);
        }

        // ---- 8x v_wmma_f32_16x16x32_bf16 (overlaps the async DMA) ----
        #pragma unroll
        for (int i = 0; i < 2; ++i)
            #pragma unroll
            for (int j = 0; j < 4; ++j)
                acc[i][j] = __builtin_amdgcn_wmma_f32_16x16x32_bf16(
                    false, af[i], false, bfr[j], (short)0, acc[i][j], false, false);

        wait_async0();       // next tile's copies (this wave) have landed
        __syncthreads();     // whole block: safe to read other buffer / reuse this one
        buf ^= 1;
    }

    // ---- epilogue: lane l15 = col, vgpr r = row (+8*half); branch-free ----
    float*  Cf = (float*)Cv;
    __bf16* Cb = (__bf16*)Cv;
    #pragma unroll
    for (int i = 0; i < 2; ++i) {
        #pragma unroll
        for (int j = 0; j < 4; ++j) {
            const int m0 = rowBase + wm * 32 + 16 * i + 8 * half;
            const int n  = colBase + wn * 64 + 16 * j + l15;
            const float bn = HB ? bias[n] : 0.f;
            #pragma unroll
            for (int r = 0; r < 8; ++r) {
                const size_t idx = (size_t)(m0 + r) * N + n;
                float val = acc[i][j][r] + bn;
                if (HR)   val += res[idx];
                if (RELU) val = fmaxf(val, 0.f);
                if (OB16) Cb[idx] = (__bf16)val;
                else      Cf[idx] = val;
            }
        }
    }
}

// ---------------------------------------------------------------------------
// Causal softmax attention: one block per (b, h, t); emits bf16 (GEMM A-side)
// ---------------------------------------------------------------------------
__global__ __launch_bounds__(256)
void attn_kernel(const float* __restrict__ Q, const float* __restrict__ Kp,
                 const float* __restrict__ Vp, __bf16* __restrict__ O)
{
    __shared__ float qv[HD];
    __shared__ float sc[TT];
    __shared__ float red[256];
    __shared__ float part[4][HD];

    const int t = blockIdx.x, h = blockIdx.y, b = blockIdx.z;
    const int tid = threadIdx.x;
    const size_t hoff = (size_t)h * HD;
    const size_t qrow = ((size_t)(b * TT + t) * DD) + hoff;

    if (tid < HD) qv[tid] = Q[qrow + tid];
    __syncthreads();

    const int nk = t + 1;

    float lmax = -INFINITY;
    for (int k0 = tid; k0 < nk; k0 += 256) {
        const float* kp = Kp + ((size_t)(b * TT + k0) * DD) + hoff;
        float dot = 0.f;
        #pragma unroll
        for (int d = 0; d < HD; ++d) dot += qv[d] * kp[d];
        dot *= 0.125f;                 // HD^-0.5
        sc[k0] = dot;
        lmax = fmaxf(lmax, dot);
    }
    red[tid] = lmax; __syncthreads();
    for (int off = 128; off > 0; off >>= 1) {
        if (tid < off) red[tid] = fmaxf(red[tid], red[tid + off]);
        __syncthreads();
    }
    const float mx = red[0];
    __syncthreads();

    float lsum = 0.f;
    for (int k0 = tid; k0 < nk; k0 += 256) {
        float e = __expf(sc[k0] - mx);
        sc[k0] = e;
        lsum += e;
    }
    red[tid] = lsum; __syncthreads();
    for (int off = 128; off > 0; off >>= 1) {
        if (tid < off) red[tid] += red[tid + off];
        __syncthreads();
    }
    const float inv = 1.f / red[0];
    __syncthreads();

    const int strip = tid >> 6;
    const int d     = tid & 63;
    float acc = 0.f;
    for (int k0 = strip; k0 < nk; k0 += 4)
        acc += sc[k0] * Vp[((size_t)(b * TT + k0) * DD) + hoff + d];
    part[strip][d] = acc;
    __syncthreads();

    if (tid < HD) {
        float o = (part[0][tid] + part[1][tid] + part[2][tid] + part[3][tid]) * inv;
        O[qrow + tid] = (__bf16)o;
    }
}

// ---------------------------------------------------------------------------
// Host orchestration
// ---------------------------------------------------------------------------
extern "C" void kernel_launch(void* const* d_in, const int* in_sizes, int n_in,
                              void* d_out, int out_size, void* d_ws, size_t ws_size,
                              hipStream_t stream)
{
    (void)in_sizes; (void)n_in; (void)out_size; (void)ws_size;

    const int*   x    = (const int*)  d_in[0];
    const float* tok  = (const float*)d_in[1];
    const float* pos  = (const float*)d_in[2];
    const float* Wq   = (const float*)d_in[3];
    const float* Wk   = (const float*)d_in[4];
    const float* Wv   = (const float*)d_in[5];
    const float* Wo   = (const float*)d_in[6];
    const float* bo   = (const float*)d_in[7];
    const float* ln1g = (const float*)d_in[8];
    const float* ln1b = (const float*)d_in[9];
    const float* W1   = (const float*)d_in[10];
    const float* b1   = (const float*)d_in[11];
    const float* W2   = (const float*)d_in[12];
    const float* b2   = (const float*)d_in[13];
    const float* ln2g = (const float*)d_in[14];
    const float* ln2b = (const float*)d_in[15];
    const float* lnfg = (const float*)d_in[16];
    const float* lnfb = (const float*)d_in[17];
    const float* Wlm  = (const float*)d_in[18];
    float* out = (float*)d_out;

    // ---- workspace carve-up ----
    char* cur = (char*)d_ws;
    auto take = [&](size_t bytes) { char* p = cur; cur += (bytes + 255) & ~(size_t)255; return p; };

    const size_t AD = (size_t)MM * DD;
    float*  hA   = (float*) take(AD * 4);
    float*  hB   = (float*) take(AD * 4);
    float*  q    = (float*) take(AD * 4);
    float*  k    = (float*) take(AD * 4);
    float*  v    = (float*) take(AD * 4);
    __bf16* a_bf = (__bf16*)take(AD * 2);
    __bf16* o_bf = (__bf16*)take(AD * 2);
    __bf16* g_bf = (__bf16*)take((size_t)MM * DFF * 2);
    __bf16* WqT  = (__bf16*)take((size_t)NBLK * DD * DD * 2);
    __bf16* WkT  = (__bf16*)take((size_t)NBLK * DD * DD * 2);
    __bf16* WvT  = (__bf16*)take((size_t)NBLK * DD * DD * 2);
    __bf16* WoT  = (__bf16*)take((size_t)NBLK * DD * DD * 2);
    __bf16* W1T  = (__bf16*)take((size_t)NBLK * DD * DFF * 2);
    __bf16* W2T  = (__bf16*)take((size_t)NBLK * DFF * DD * 2);
    __bf16* WlmT = (__bf16*)take((size_t)DD * VV * 2);

    const dim3 blk(256);

    // ---- weight prep: transpose + bf16 convert ----
    auto tc = [&](const float* src, __bf16* dst, int Kd, int Nd) {
        transpose_cvt_kernel<<<dim3(Nd / 32, Kd / 32), blk, 0, stream>>>(src, dst, Kd, Nd);
    };
    for (int i = 0; i < NBLK; ++i) {
        tc(Wq + (size_t)i * DD * DD,  WqT + (size_t)i * DD * DD,  DD, DD);
        tc(Wk + (size_t)i * DD * DD,  WkT + (size_t)i * DD * DD,  DD, DD);
        tc(Wv + (size_t)i * DD * DD,  WvT + (size_t)i * DD * DD,  DD, DD);
        tc(Wo + (size_t)i * DD * DD,  WoT + (size_t)i * DD * DD,  DD, DD);
        tc(W1 + (size_t)i * DD * DFF, W1T + (size_t)i * DD * DFF, DD, DFF);
        tc(W2 + (size_t)i * DFF * DD, W2T + (size_t)i * DFF * DD, DFF, DD);
    }
    tc(Wlm, WlmT, DD, VV);

    const dim3 gD(DD  / BN, MM / BM);            // 8  x 32
    const dim3 gF(DFF / BN, MM / BM);            // 32 x 32
    const dim3 gV(VV  / BN, MM / BM);            // 250 x 32

    embed_kernel<<<dim3((unsigned)(((size_t)MM * DD + 255) / 256)), blk, 0, stream>>>(
        x, tok, pos, hA);

    for (int i = 0; i < NBLK; ++i) {
        const __bf16* WqT_i = WqT + (size_t)i * DD * DD;
        const __bf16* WkT_i = WkT + (size_t)i * DD * DD;
        const __bf16* WvT_i = WvT + (size_t)i * DD * DD;
        const __bf16* WoT_i = WoT + (size_t)i * DD * DD;
        const __bf16* W1T_i = W1T + (size_t)i * DD * DFF;
        const __bf16* W2T_i = W2T + (size_t)i * DFF * DD;
        const float*  bo_i  = bo + (size_t)i * DD;
        const float*  bb1   = b1 + (size_t)i * DFF;
        const float*  bb2   = b2 + (size_t)i * DD;

        layernorm_kernel<<<MM, blk, 0, stream>>>(hA, ln1g + (size_t)i * DD, ln1b + (size_t)i * DD, a_bf);
        gemm_bf16_wmma<false,false,false,false><<<gD, blk, 0, stream>>>(
            a_bf, WqT_i, nullptr, nullptr, q, MM, DD, DD);
        gemm_bf16_wmma<false,false,false,false><<<gD, blk, 0, stream>>>(
            a_bf, WkT_i, nullptr, nullptr, k, MM, DD, DD);
        gemm_bf16_wmma<false,false,false,false><<<gD, blk, 0, stream>>>(
            a_bf, WvT_i, nullptr, nullptr, v, MM, DD, DD);
        attn_kernel<<<dim3(TT, HH, BB), blk, 0, stream>>>(q, k, v, o_bf);
        gemm_bf16_wmma<true,true,false,false><<<gD, blk, 0, stream>>>(
            o_bf, WoT_i, bo_i, hA, hB, MM, DD, DD);
        layernorm_kernel<<<MM, blk, 0, stream>>>(hB, ln2g + (size_t)i * DD, ln2b + (size_t)i * DD, a_bf);
        gemm_bf16_wmma<true,false,true,true><<<gF, blk, 0, stream>>>(
            a_bf, W1T_i, bb1, nullptr, g_bf, MM, DFF, DD);
        gemm_bf16_wmma<true,true,false,false><<<gD, blk, 0, stream>>>(
            g_bf, W2T_i, bb2, hB, hA, MM, DD, DFF);
    }

    layernorm_kernel<<<MM, blk, 0, stream>>>(hA, lnfg, lnfb, a_bf);
    gemm_bf16_wmma<false,false,false,false><<<gV, blk, 0, stream>>>(
        a_bf, WlmT, nullptr, nullptr, out, MM, VV, DD);
}
